// EncoderLayer_15779709845770
// MI455X (gfx1250) — matrix operand order repaired and checked by reading, not compiled, not run on hardware
//
#include <hip/hip_runtime.h>

#define D_MODEL 1024
#define SEQ     1024
#define BATCH   8
#define HEADS   16
#define DHEAD   64
#define DFF     2048
#define NTOK    (BATCH * SEQ)   // 8192 tokens

typedef __attribute__((ext_vector_type(16))) _Float16 v16h;
typedef __attribute__((ext_vector_type(8)))  float    v8f;
typedef __attribute__((ext_vector_type(8)))  _Float16 h8;
typedef __attribute__((ext_vector_type(4)))  int      v4i;

// CDNA5 async global->LDS path (ASYNCcnt), guarded so either toolchain compiles.
#if defined(__gfx1250__) && __has_builtin(__builtin_amdgcn_global_load_async_to_lds_b128)
#define ASYNC_OK 1
// clang-22 signature: (v4i __attribute__((address_space(1)))*,  <- global src
//                      v4i __attribute__((address_space(3)))*,  <- LDS dst
//                      imm offset, imm cpol)
typedef __attribute__((address_space(1))) v4i* g_v4i_ptr;
typedef __attribute__((address_space(3))) v4i* l_v4i_ptr;
#else
#define ASYNC_OK 0
#endif

static __device__ __forceinline__ int lane_id() { return threadIdx.x & 31; }

// 16-byte global -> LDS copy; async (tracked by ASYNCcnt) when available.
static __device__ __forceinline__ void copy16_g2l(_Float16* lds, const _Float16* g) {
#if ASYNC_OK
  __builtin_amdgcn_global_load_async_to_lds_b128(
      (g_v4i_ptr)g, (l_v4i_ptr)lds, 0, 0);
#else
  *(h8*)lds = *(const h8*)g;
#endif
}

template <int N>
static __device__ __forceinline__ void wait_async() {
#if ASYNC_OK
#if __has_builtin(__builtin_amdgcn_s_wait_asynccnt)
  __builtin_amdgcn_s_wait_asynccnt(N);
#else
  asm volatile("s_wait_asynccnt %0" ::"i"(N) : "memory");
#endif
#endif
}

// --------------------------------------------------------------------------
// WMMA fragment loaders (wave32, v_wmma_f32_16x16x32_f16 layouts).
//
// A (16x32, row-major): lane<16 row=lane, K = k0+{0..7,16..23};
//                       lane>=16 row=lane-16, K = k0+{8..15,24..31}.
// B (32x16) from B^T (NxK row-major): lane<16 col=lane, K=k0+{0..7,16..23};
//                                     lane>=16 col=lane-16, K=k0+{8..15,24..31}.
static __device__ __forceinline__ v16h load_a_frag(const _Float16* __restrict__ A,
                                                   int lda, int m0, int k0) {
  const int lane = lane_id();
  const _Float16* p = A + (size_t)(m0 + (lane & 15)) * lda + k0 + (lane >> 4) * 8;
  h8 lo = *(const h8*)(p);
  h8 hi = *(const h8*)(p + 16);
  v16h r;
#pragma unroll
  for (int i = 0; i < 8; ++i) { r[i] = lo[i]; r[8 + i] = hi[i]; }
  return r;
}

static __device__ __forceinline__ v16h load_bt_frag(const _Float16* __restrict__ Bt,
                                                    int ldb, int n0, int k0) {
  const int lane = lane_id();
  const _Float16* p = Bt + (size_t)(n0 + (lane & 15)) * ldb + k0 + (lane >> 4) * 8;
  h8 lo = *(const h8*)(p);
  h8 hi = *(const h8*)(p + 16);
  v16h r;
#pragma unroll
  for (int i = 0; i < 8; ++i) { r[i] = lo[i]; r[8 + i] = hi[i]; }
  return r;
}

// LDS-tile fragment loaders; tile rows are LDSTRIDE f16 apart (80B: the
// 20*n mod 64 bank rotation is a perfect 16-lane permutation -> conflict-free).
#define LDSTRIDE 40

static __device__ __forceinline__ v16h lds_a_frag(const _Float16* base) {
  const int lane = lane_id();
  const _Float16* p = base + (size_t)(lane & 15) * LDSTRIDE + (lane >> 4) * 8;
  h8 lo = *(const h8*)(p);
  h8 hi = *(const h8*)(p + 16);
  v16h r;
#pragma unroll
  for (int i = 0; i < 8; ++i) { r[i] = lo[i]; r[8 + i] = hi[i]; }
  return r;
}

static __device__ __forceinline__ v16h lds_b_frag(const _Float16* base) {
  const int lane = lane_id();
  const _Float16* p = base + (size_t)(lane & 15) * LDSTRIDE + (lane >> 4) * 8;
  h8 lo = *(const h8*)(p);
  h8 hi = *(const h8*)(p + 16);
  v16h r;
#pragma unroll
  for (int i = 0; i < 8; ++i) { r[i] = lo[i]; r[8 + i] = hi[i]; }
  return r;
}

static __device__ __forceinline__ v8f wmma_f16(v16h a, v16h b, v8f c) {
  return __builtin_amdgcn_wmma_f32_16x16x32_f16(false, a, false, b,
                                                (short)0, c, false, false);
}

// --------------------------------------------------------------------------
// GEMM: C[M,N] = act(A[M,K] @ B[K,N] + bias) + residual.
// A f16 row-major (lda=K); Bt = B^T f16 (N x K). 256 threads = 8 waves.
// Block tile 128x128; waves 4(M) x 2(N); wave computes 32x64 (8 acc tiles).
// A/B K-slices (128x32) double-buffered in LDS via async global->LDS loads.
__global__ __launch_bounds__(256) void gemm_f16(
    const _Float16* __restrict__ A, const _Float16* __restrict__ Bt,
    const float* __restrict__ bias, const float* __restrict__ residual,
    float* __restrict__ outF, _Float16* __restrict__ outH,
    int M, int N, int K, int do_relu)
{
  __shared__ __align__(16) _Float16 Abuf[2][128][LDSTRIDE];
  __shared__ __align__(16) _Float16 Bbuf[2][128][LDSTRIDE];

  const int wave = threadIdx.x >> 5;
  const int lane = lane_id();
  const int block_m = blockIdx.y * 128;
  const int block_n = blockIdx.x * 128;
  const int mw = (wave >> 1) * 32;   // wave's rows within block
  const int nw = (wave & 1) * 64;    // wave's cols within block

  // Stage one 128x32 K-slice of A and of Bt into LDS buffer `sbuf`.
  // 512 16B chunks per matrix; each of 256 threads copies 2 A + 2 B chunks
  // (4 ASYNCcnt increments per wave per stage).
  auto stage = [&](int sbuf, int k0) {
#pragma unroll
    for (int j = 0; j < 2; ++j) {
      const int c   = (int)threadIdx.x + j * 256;
      const int row = c >> 2;
      const int kc  = (c & 3) * 8;
      copy16_g2l(&Abuf[sbuf][row][kc], A  + (size_t)(block_m + row) * K + k0 + kc);
      copy16_g2l(&Bbuf[sbuf][row][kc], Bt + (size_t)(block_n + row) * K + k0 + kc);
    }
  };

  v8f acc0[4] = {};
  v8f acc1[4] = {};

  stage(0, 0);
  int buf = 0;
  for (int k0 = 0; k0 < K; k0 += 32, buf ^= 1) {
    const bool has_next = (k0 + 32) < K;
    if (has_next) {
      stage(buf ^ 1, k0 + 32);
      wait_async<4>();   // newest 4 in flight; the 4 for `buf` are complete
    } else {
      wait_async<0>();
    }
    __syncthreads();     // all waves' chunks for `buf` visible

    v16h a0 = lds_a_frag(&Abuf[buf][mw][0]);
    v16h a1 = lds_a_frag(&Abuf[buf][mw + 16][0]);
#pragma unroll
    for (int t = 0; t < 4; ++t) {
      v16h b = lds_b_frag(&Bbuf[buf][nw + t * 16][0]);
      acc0[t] = wmma_f16(a0, b, acc0[t]);
      acc1[t] = wmma_f16(a1, b, acc1[t]);
    }
    __syncthreads();     // done reading `buf` before it is restaged
  }

  const int mbase = (lane >> 4) * 8;  // C layout: lanes 0-15 rows 0-7, 16-31 rows 8-15
  const int ncol  = lane & 15;
#pragma unroll
  for (int t = 0; t < 4; ++t) {
    const int col = block_n + nw + t * 16 + ncol;
    const float bv = bias[col];
#pragma unroll
    for (int r = 0; r < 8; ++r) {
#pragma unroll
      for (int mt = 0; mt < 2; ++mt) {
        const int row = block_m + mw + mt * 16 + mbase + r;
        float v = (mt ? acc1[t][r] : acc0[t][r]) + bv;
        if (do_relu) v = fmaxf(v, 0.0f);
        if (residual) v += residual[(size_t)row * N + col];
        if (outF) outF[(size_t)row * N + col] = v;
        else      outH[(size_t)row * N + col] = (_Float16)v;
      }
    }
  }
}

// --------------------------------------------------------------------------
// LayerNorm (ddof=1, std + 1e-6) -> f16. One 256-thread block per row.
__global__ __launch_bounds__(256) void layernorm_to_f16(
    const float* __restrict__ x, const float* __restrict__ gain,
    const float* __restrict__ beta, _Float16* __restrict__ out)
{
  const int row = blockIdx.x;
  const float* xr = x + (size_t)row * D_MODEL;
  float v[4], s = 0.f, sq = 0.f;
#pragma unroll
  for (int j = 0; j < 4; ++j) {
    v[j] = xr[threadIdx.x + j * 256];
    s += v[j]; sq += v[j] * v[j];
  }
  __shared__ float r1[256], r2[256];
  r1[threadIdx.x] = s; r2[threadIdx.x] = sq;
  __syncthreads();
  for (int off = 128; off > 0; off >>= 1) {
    if ((int)threadIdx.x < off) {
      r1[threadIdx.x] += r1[threadIdx.x + off];
      r2[threadIdx.x] += r2[threadIdx.x + off];
    }
    __syncthreads();
  }
  const float mean = r1[0] / D_MODEL;
  const float var  = (r2[0] - mean * r1[0]) / (D_MODEL - 1);
  const float istd = 1.0f / (sqrtf(var) + 1e-6f);
  _Float16* orow = out + (size_t)row * D_MODEL;
#pragma unroll
  for (int j = 0; j < 4; ++j) {
    const int c = threadIdx.x + j * 256;
    orow[c] = (_Float16)(gain[c] * istd * (v[j] - mean) + beta[c]);
  }
}

// --------------------------------------------------------------------------
// Transpose + f32->f16 convert: out[c*R + r] = (f16) in[r*C + c]. block (32,8).
__global__ __launch_bounds__(256) void transpose_f32_to_f16(
    const float* __restrict__ in, _Float16* __restrict__ out, int R, int C)
{
  __shared__ float tile[32][33];
  const int c0 = blockIdx.x * 32, r0 = blockIdx.y * 32;
#pragma unroll
  for (int j = 0; j < 4; ++j) {
    const int r = r0 + threadIdx.y + j * 8;
    tile[threadIdx.y + j * 8][threadIdx.x] = in[(size_t)r * C + c0 + threadIdx.x];
  }
  __syncthreads();
#pragma unroll
  for (int j = 0; j < 4; ++j) {
    const int c = c0 + threadIdx.y + j * 8;
    out[(size_t)c * R + r0 + threadIdx.x] = (_Float16)tile[threadIdx.x][threadIdx.y + j * 8];
  }
}

// Transpose per-(b,h) V slab: vT[bh][d][s] = v[(b*S+s)*D + h*64 + d]. block (32,8).
__global__ __launch_bounds__(256) void transpose_heads_f16(
    const _Float16* __restrict__ v, _Float16* __restrict__ vT)
{
  __shared__ _Float16 tile[32][34];
  const int bh = blockIdx.z;
  const int b = bh >> 4, h = bh & 15;
  const int d0 = blockIdx.x * 32, s0 = blockIdx.y * 32;
  const _Float16* src = v + (size_t)b * SEQ * D_MODEL + h * DHEAD;
#pragma unroll
  for (int j = 0; j < 4; ++j) {
    const int srow = s0 + threadIdx.y + j * 8;
    tile[threadIdx.y + j * 8][threadIdx.x] = src[(size_t)srow * D_MODEL + d0 + threadIdx.x];
  }
  __syncthreads();
  _Float16* dst = vT + (size_t)bh * DHEAD * SEQ;
#pragma unroll
  for (int j = 0; j < 4; ++j) {
    const int drow = d0 + threadIdx.y + j * 8;
    dst[(size_t)drow * SEQ + s0 + threadIdx.x] = tile[threadIdx.x][threadIdx.y + j * 8];
  }
}

// --------------------------------------------------------------------------
// Flash attention: grid (S/64, B*H), 128 threads = 4 waves.
// Each wave owns 16 query rows; online softmax over key blocks of 32.
__global__ __launch_bounds__(128) void flash_attn(
    const _Float16* __restrict__ q, const _Float16* __restrict__ kmat,
    const _Float16* __restrict__ vT, const int* __restrict__ mask,
    _Float16* __restrict__ o)
{
  const int wave = threadIdx.x >> 5;
  const int lane = lane_id();
  const int bh = blockIdx.y;
  const int b = bh >> 4, h = bh & 15;
  const int q0 = blockIdx.x * 64 + wave * 16;

  const _Float16* qb = q    + (size_t)b * SEQ * D_MODEL + h * DHEAD;
  const _Float16* kb = kmat + (size_t)b * SEQ * D_MODEL + h * DHEAD;
  const _Float16* vb = vT   + (size_t)bh * DHEAD * SEQ;
  const int* mb = mask + b * SEQ;

  const v16h aq0 = load_a_frag(qb, D_MODEL, q0, 0);
  const v16h aq1 = load_a_frag(qb, D_MODEL, q0, 32);

  __shared__ __align__(16) _Float16 pt[4][16][32];  // per-wave P staging tile

  float mrow[8], lrow[8], corr[8];
  v8f acc[4] = {};
#pragma unroll
  for (int r = 0; r < 8; ++r) { mrow[r] = -1e30f; lrow[r] = 0.f; }

  const int hlf = lane >> 4;
  const int nc  = lane & 15;

  for (int j = 0; j < SEQ / 32; ++j) {
    const int k0 = j * 32;
    v8f s0 = {}, s1 = {};
    {
      v16h b0 = load_bt_frag(kb, D_MODEL, k0,      0);
      v16h b1 = load_bt_frag(kb, D_MODEL, k0,      32);
      s0 = wmma_f16(aq0, b0, s0);
      s0 = wmma_f16(aq1, b1, s0);
      v16h b2 = load_bt_frag(kb, D_MODEL, k0 + 16, 0);
      v16h b3 = load_bt_frag(kb, D_MODEL, k0 + 16, 32);
      s1 = wmma_f16(aq0, b2, s1);
      s1 = wmma_f16(aq1, b3, s1);
    }
    const int mv0 = mb[k0 + nc];
    const int mv1 = mb[k0 + 16 + nc];
    float p0[8], p1[8];
#pragma unroll
    for (int r = 0; r < 8; ++r) {
      p0[r] = mv0 ? s0[r] * 0.125f : -1e9f;   // 1/sqrt(64)
      p1[r] = mv1 ? s1[r] * 0.125f : -1e9f;
    }
#pragma unroll
    for (int r = 0; r < 8; ++r) {
      float t = fmaxf(p0[r], p1[r]);
#pragma unroll
      for (int off = 8; off > 0; off >>= 1) t = fmaxf(t, __shfl_xor(t, off, 32));
      const float mnew = fmaxf(mrow[r], t);
      corr[r] = __expf(mrow[r] - mnew);
      mrow[r] = mnew;
      p0[r] = __expf(p0[r] - mnew);
      p1[r] = __expf(p1[r] - mnew);
      float rs = p0[r] + p1[r];
#pragma unroll
      for (int off = 8; off > 0; off >>= 1) rs += __shfl_xor(rs, off, 32);
      lrow[r] = lrow[r] * corr[r] + rs;
    }
#pragma unroll
    for (int t4 = 0; t4 < 4; ++t4)
#pragma unroll
      for (int r = 0; r < 8; ++r) acc[t4][r] *= corr[r];

    // C-layout P -> LDS -> A-fragment layout (same-wave DS ops are in order).
#pragma unroll
    for (int r = 0; r < 8; ++r) {
      const int m = r + 8 * hlf;
      pt[wave][m][nc]      = (_Float16)p0[r];
      pt[wave][m][16 + nc] = (_Float16)p1[r];
    }
    v16h ap;
    {
      const _Float16* pr = &pt[wave][nc][hlf * 8];
      h8 lo = *(const h8*)(pr);
      h8 hi = *(const h8*)(pr + 16);
#pragma unroll
      for (int i = 0; i < 8; ++i) { ap[i] = lo[i]; ap[8 + i] = hi[i]; }
    }
#pragma unroll
    for (int t4 = 0; t4 < 4; ++t4) {
      v16h bvf = load_bt_frag(vb, SEQ, t4 * 16, k0);
      acc[t4] = wmma_f16(ap, bvf, acc[t4]);
    }
  }

  _Float16* ob = o + (size_t)b * SEQ * D_MODEL + h * DHEAD;
#pragma unroll
  for (int t4 = 0; t4 < 4; ++t4)
#pragma unroll
    for (int r = 0; r < 8; ++r) {
      const int row = q0 + r + 8 * hlf;
      ob[(size_t)row * D_MODEL + t4 * 16 + nc] = (_Float16)(acc[t4][r] / lrow[r]);
    }
}

// --------------------------------------------------------------------------
extern "C" void kernel_launch(void* const* d_in, const int* in_sizes, int n_in,
                              void* d_out, int out_size, void* d_ws, size_t ws_size,
                              hipStream_t stream)
{
  (void)in_sizes; (void)n_in; (void)out_size; (void)ws_size;
  const float* x   = (const float*)d_in[0];
  const int*   msk = (const int*)  d_in[1];
  const float* wq  = (const float*)d_in[2];
  const float* bq  = (const float*)d_in[3];
  const float* wk  = (const float*)d_in[4];
  const float* bk  = (const float*)d_in[5];
  const float* wv  = (const float*)d_in[6];
  const float* bv  = (const float*)d_in[7];
  const float* wo  = (const float*)d_in[8];
  const float* bo  = (const float*)d_in[9];
  const float* w1  = (const float*)d_in[10];
  const float* b1  = (const float*)d_in[11];
  const float* w2  = (const float*)d_in[12];
  const float* b2  = (const float*)d_in[13];
  const float* g1  = (const float*)d_in[14];
  const float* be1 = (const float*)d_in[15];
  const float* g2  = (const float*)d_in[16];
  const float* be2 = (const float*)d_in[17];

  float* outp = (float*)d_out;                         // tuple elem 0: out
  float* attn = outp + (size_t)NTOK * D_MODEL;         // tuple elem 1: attn

  char* w = (char*)d_ws;
  const size_t MB = (size_t)1 << 20;
  _Float16* xn  = (_Float16*)(w + 0   * MB);   // 16 MB
  _Float16* qh  = (_Float16*)(w + 16  * MB);   // 16 MB
  _Float16* kh  = (_Float16*)(w + 32  * MB);   // 16 MB
  _Float16* vh  = (_Float16*)(w + 48  * MB);   // 16 MB
  _Float16* vTt = (_Float16*)(w + 64  * MB);   // 16 MB
  _Float16* oh  = (_Float16*)(w + 80  * MB);   // 16 MB
  _Float16* an  = (_Float16*)(w + 96  * MB);   // 16 MB
  _Float16* h1  = (_Float16*)(w + 112 * MB);   // 32 MB
  _Float16* wqT = (_Float16*)(w + 144 * MB);   // 2 MB each
  _Float16* wkT = (_Float16*)(w + 146 * MB);
  _Float16* wvT = (_Float16*)(w + 148 * MB);
  _Float16* woT = (_Float16*)(w + 150 * MB);
  _Float16* w1T = (_Float16*)(w + 152 * MB);   // 4 MB
  _Float16* w2T = (_Float16*)(w + 156 * MB);   // 4 MB

  const dim3 tb(32, 8, 1);
  transpose_f32_to_f16<<<dim3(32, 32), tb, 0, stream>>>(wq, wqT, D_MODEL, D_MODEL);
  transpose_f32_to_f16<<<dim3(32, 32), tb, 0, stream>>>(wk, wkT, D_MODEL, D_MODEL);
  transpose_f32_to_f16<<<dim3(32, 32), tb, 0, stream>>>(wv, wvT, D_MODEL, D_MODEL);
  transpose_f32_to_f16<<<dim3(32, 32), tb, 0, stream>>>(wo, woT, D_MODEL, D_MODEL);
  transpose_f32_to_f16<<<dim3(64, 32), tb, 0, stream>>>(w1, w1T, D_MODEL, DFF);
  transpose_f32_to_f16<<<dim3(32, 64), tb, 0, stream>>>(w2, w2T, DFF, D_MODEL);

  layernorm_to_f16<<<NTOK, 256, 0, stream>>>(x, g1, be1, xn);

  gemm_f16<<<dim3(D_MODEL / 128, NTOK / 128), 256, 0, stream>>>(
      xn, wqT, bq, nullptr, nullptr, qh, NTOK, D_MODEL, D_MODEL, 0);
  gemm_f16<<<dim3(D_MODEL / 128, NTOK / 128), 256, 0, stream>>>(
      xn, wkT, bk, nullptr, nullptr, kh, NTOK, D_MODEL, D_MODEL, 0);
  gemm_f16<<<dim3(D_MODEL / 128, NTOK / 128), 256, 0, stream>>>(
      xn, wvT, bv, nullptr, nullptr, vh, NTOK, D_MODEL, D_MODEL, 0);

  transpose_heads_f16<<<dim3(DHEAD / 32, SEQ / 32, BATCH * HEADS), tb, 0, stream>>>(vh, vTt);

  flash_attn<<<dim3(SEQ / 64, BATCH * HEADS), 128, 0, stream>>>(qh, kh, vTt, msk, oh);

  // attn = x + o @ wo + bo   (written straight into d_out's second tensor)
  gemm_f16<<<dim3(D_MODEL / 128, NTOK / 128), 256, 0, stream>>>(
      oh, woT, bo, x, attn, nullptr, NTOK, D_MODEL, D_MODEL, 0);

  layernorm_to_f16<<<NTOK, 256, 0, stream>>>(attn, g2, be2, an);

  gemm_f16<<<dim3(DFF / 128, NTOK / 128), 256, 0, stream>>>(
      an, w1T, b1, nullptr, nullptr, h1, NTOK, DFF, D_MODEL, 1);

  // out = attn + relu(an@w1+b1) @ w2 + b2
  gemm_f16<<<dim3(D_MODEL / 128, NTOK / 128), 256, 0, stream>>>(
      h1, w2T, b2, attn, outp, nullptr, NTOK, D_MODEL, DFF, 0);
}
